// MRFM_14173392076936
// MI455X (gfx1250) — matrix operand order repaired and checked by reading, not compiled, not run on hardware
//
#include <hip/hip_runtime.h>
#include <hip/hip_bf16.h>
#include <math.h>

typedef __attribute__((ext_vector_type(16))) _Float16 v16h;
typedef __attribute__((ext_vector_type(8)))  float    v8f;

#define BATCH 8
#define DE 64
#define DI 128
#define HH 160
#define WWD 160
#define HWP (HH*WWD)     // 25600
#define NPT (HWP/16)     // 1600
#define HEADS 4
#define DK 32
#define NCHUNK 50        // 50 * 512 = 25600
#define CHUNKP 512
#define CTXSZ (BATCH*HEADS*DK*DK)   // 32768

// ---------------------------------------------------------------------------
// 1x1-conv (channel GEMM) via WMMA f32_16x16x32_f16, compile-time shapes.
// Y[n][o][p] = sum_c W[o][c]*X[n][c][p] (+bias)(ReLU?)(+resid)
// X = concat(X1[C1], X2[CIN-C1]).  One wave: one 16-pixel tile x ALL output
// channel tiles (B fragment reused across COUT/16 WMMAs).
// ---------------------------------------------------------------------------
template<int CIN, int COUT, int C1>
__global__ __launch_bounds__(128) void conv1x1_wmma(
    const float* __restrict__ X1, const float* __restrict__ X2,
    const float* __restrict__ Wt, const float* __restrict__ bias,
    const float* __restrict__ resid, int relu, float* __restrict__ Y)
{
  constexpr int NT = (COUT + 15) / 16;
  long gwave = ((long)blockIdx.x * blockDim.x + threadIdx.x) >> 5;
  if (gwave >= (long)BATCH * NPT) return;
  int lane = threadIdx.x & 31;
  int pt = (int)(gwave % NPT);
  int n  = (int)(gwave / NPT);
  int lrow = lane & 15, lhi = lane >> 4;
  int p = pt * 16 + lrow;

  v8f zero = {};
  v8f acc[NT];
#pragma unroll
  for (int t = 0; t < NT; ++t) acc[t] = zero;

#pragma unroll
  for (int k0 = 0; k0 < CIN; k0 += 32) {
    // ---- B fragment (activations), 16-bit 32x16 layout -----------------
    const float* Xs;
    int coff;
    if (k0 < C1) { Xs = X1 + (size_t)n * C1 * HWP;          coff = k0; }
    else         { Xs = X2 + (size_t)n * (CIN - C1) * HWP;  coff = k0 - C1; }
    v16h b;
#pragma unroll
    for (int j = 0; j < 8; ++j) {
      int kb = coff + lhi * 16 + j * 2;
      float v0, v1;
      if constexpr (CIN % 32 != 0) {
        int kabs = k0 + lhi * 16 + j * 2;
        v0 = (kabs     < CIN) ? Xs[(size_t)kb * HWP + p]       : 0.f;
        v1 = (kabs + 1 < CIN) ? Xs[(size_t)(kb + 1) * HWP + p] : 0.f;
      } else {
        v0 = Xs[(size_t)kb * HWP + p];
        v1 = Xs[(size_t)(kb + 1) * HWP + p];
      }
      b[2 * j]     = (_Float16)v0;
      b[2 * j + 1] = (_Float16)v1;
    }
    // ---- A fragments (weights) + WMMA per output tile ------------------
#pragma unroll
    for (int t = 0; t < NT; ++t) {
      int m = t * 16 + lrow;
      v16h a;
#pragma unroll
      for (int j = 0; j < 8; ++j) {
        int ka = k0 + lhi * 16 + j * 2;
        float w0, w1;
        if constexpr (COUT % 16 != 0 || CIN % 32 != 0) {
          bool mok = (COUT % 16 == 0) || (m < COUT);
          w0 = (mok && ((CIN % 32 == 0) || (ka     < CIN))) ? Wt[(size_t)m * CIN + ka]     : 0.f;
          w1 = (mok && ((CIN % 32 == 0) || (ka + 1 < CIN))) ? Wt[(size_t)m * CIN + ka + 1] : 0.f;
        } else {
          float2 wv = *(const float2*)(Wt + (size_t)m * CIN + ka);
          w0 = wv.x; w1 = wv.y;
        }
        a[2 * j]     = (_Float16)w0;
        a[2 * j + 1] = (_Float16)w1;
      }
      acc[t] = __builtin_amdgcn_wmma_f32_16x16x32_f16(false, a, false, b,
                                                      (short)0, acc[t], false, false);
    }
  }
  // ---- epilogue ---------------------------------------------------------
#pragma unroll
  for (int t = 0; t < NT; ++t) {
#pragma unroll
    for (int j = 0; j < 8; ++j) {
      int o = t * 16 + j + lhi * 8;
      if (COUT % 16 == 0 || o < COUT) {
        float v = acc[t][j];
        if (bias) v += bias[o];
        if (relu) v = fmaxf(v, 0.f);
        size_t idx = ((size_t)n * COUT + o) * HWP + p;
        if (resid) v += resid[idx];
        Y[idx] = v;
      }
    }
  }
}

// ---------------------------------------------------------------------------
// part[chunk][n,h,k,v] = sum_{p in chunk} Ksoft[n,h*32+k,p] * V[n,h*32+v,p]
// One wave: (n,h,chunk of 512 px); both k-tiles x both v-tiles fused
// (K/V rows each read once, 4 WMMAs per iteration). Plain stores ->
// deterministic second-stage reduction (no float atomics).
// ---------------------------------------------------------------------------
__global__ __launch_bounds__(128) void ctx_wmma(
    const float* __restrict__ Ks, const float* __restrict__ V,
    float* __restrict__ part)
{
  long gwave = ((long)blockIdx.x * blockDim.x + threadIdx.x) >> 5;
  if (gwave >= (long)BATCH * HEADS * NCHUNK) return;
  int lane = threadIdx.x & 31;
  int chunk = (int)(gwave % NCHUNK);
  int h  = (int)((gwave / NCHUNK) % HEADS);
  int n  = (int)(gwave / ((long)NCHUNK * HEADS));
  int lrow = lane & 15, lhi = lane >> 4;
  const float* kbase = Ks + ((size_t)n * DI + h * DK) * HWP;
  const float* vbase = V  + ((size_t)n * DI + h * DK) * HWP;
  int p0 = chunk * CHUNKP;
  v8f zero = {};
  v8f acc[2][2];
  acc[0][0] = zero; acc[0][1] = zero; acc[1][0] = zero; acc[1][1] = zero;
  for (int it = 0; it < CHUNKP / 32; ++it) {
    int pb = p0 + it * 32;
    v16h aK[2], bV[2];
#pragma unroll
    for (int kt = 0; kt < 2; ++kt) {
      const float* krow = kbase + (size_t)(kt * 16 + lrow) * HWP;
#pragma unroll
      for (int j = 0; j < 8; ++j) {
        int pa = pb + lhi * 16 + j * 2;
        float2 kv = *(const float2*)(krow + pa);
        aK[kt][2 * j]     = (_Float16)kv.x;
        aK[kt][2 * j + 1] = (_Float16)kv.y;
      }
    }
#pragma unroll
    for (int vt = 0; vt < 2; ++vt) {
      const float* vrow = vbase + (size_t)(vt * 16 + lrow) * HWP;
#pragma unroll
      for (int j = 0; j < 8; ++j) {
        int pv = pb + lhi * 16 + j * 2;
        float2 vv = *(const float2*)(vrow + pv);
        bV[vt][2 * j]     = (_Float16)vv.x;
        bV[vt][2 * j + 1] = (_Float16)vv.y;
      }
    }
#pragma unroll
    for (int kt = 0; kt < 2; ++kt)
#pragma unroll
      for (int vt = 0; vt < 2; ++vt)
        acc[kt][vt] = __builtin_amdgcn_wmma_f32_16x16x32_f16(
            false, aK[kt], false, bV[vt], (short)0, acc[kt][vt], false, false);
  }
  float* slot = part + (size_t)chunk * CTXSZ + (((size_t)n * HEADS + h) * DK) * DK;
#pragma unroll
  for (int kt = 0; kt < 2; ++kt)
#pragma unroll
    for (int vt = 0; vt < 2; ++vt)
#pragma unroll
      for (int j = 0; j < 8; ++j) {
        int k = kt * 16 + j + lhi * 8;
        slot[(size_t)k * DK + vt * 16 + lrow] = acc[kt][vt][j];
      }
}

// Deterministic chunk reduction; stores TRANSPOSED: ctxT[n,h,v,k].
__global__ void ctx_reduce(const float* __restrict__ part, float* __restrict__ ctxT)
{
  int idx = blockIdx.x * blockDim.x + threadIdx.x;     // over CTXSZ
  if (idx >= CTXSZ) return;
  float s = 0.f;
  for (int c = 0; c < NCHUNK; ++c) s += part[(size_t)c * CTXSZ + idx];
  int v  = idx & (DK - 1);
  int k  = (idx >> 5) & (DK - 1);
  int nh = idx >> 10;
  ctxT[((size_t)nh * DK + v) * DK + k] = s;
}

// ---------------------------------------------------------------------------
// att[n,h*32+v,p] = sum_k ctxT[n,h,v,k] * Q[n,h*32+k,p]
// One wave: (n,h,pixel tile); both v-tiles fused sharing the Q fragment.
// ---------------------------------------------------------------------------
__global__ __launch_bounds__(128) void att_wmma(
    const float* __restrict__ ctxT, const float* __restrict__ Q,
    float* __restrict__ A)
{
  long gwave = ((long)blockIdx.x * blockDim.x + threadIdx.x) >> 5;
  if (gwave >= (long)BATCH * HEADS * NPT) return;
  int lane = threadIdx.x & 31;
  int pt = (int)(gwave % NPT);
  int h  = (int)((gwave / NPT) % HEADS);
  int n  = (int)(gwave / ((long)NPT * HEADS));
  int lrow = lane & 15, lhi = lane >> 4;
  const float* cbase = ctxT + (((size_t)n * HEADS + h) * DK) * DK;   // [v][k]
  const float* qbase = Q + ((size_t)n * DI + h * DK) * HWP;
  int p = pt * 16 + lrow;
  v16h b;
#pragma unroll
  for (int j = 0; j < 8; ++j) {
    int kb = lhi * 16 + j * 2;
    b[2 * j]     = (_Float16)qbase[(size_t)kb * HWP + p];
    b[2 * j + 1] = (_Float16)qbase[(size_t)(kb + 1) * HWP + p];
  }
  v8f zero = {};
#pragma unroll
  for (int vt = 0; vt < 2; ++vt) {
    const float* crow = cbase + (size_t)(vt * 16 + lrow) * DK;
    v16h a;
#pragma unroll
    for (int j = 0; j < 8; ++j) {
      int ka = lhi * 16 + j * 2;
      float2 cv = *(const float2*)(crow + ka);
      a[2 * j]     = (_Float16)cv.x;
      a[2 * j + 1] = (_Float16)cv.y;
    }
    v8f acc = __builtin_amdgcn_wmma_f32_16x16x32_f16(false, a, false, b,
                                                     (short)0, zero, false, false);
#pragma unroll
    for (int j = 0; j < 8; ++j) {
      int c = h * DK + vt * 16 + j + lhi * 8;
      A[((size_t)n * DI + c) * HWP + p] = acc[j];
    }
  }
}

// ---------------------------------------------------------------------------
// per-(n,c) mean & max over spatial for mm = concat(ev, img)
// ---------------------------------------------------------------------------
__global__ void channel_stats(const float* __restrict__ ev,
                              const float* __restrict__ img,
                              float* __restrict__ avg, float* __restrict__ mx)
{
  int row = blockIdx.x;                 // n*(DE+DI)+c
  int n = row / (DE + DI), c = row % (DE + DI);
  const float* base = (c < DE) ? ev + ((size_t)n * DE + c) * HWP
                               : img + ((size_t)n * DI + (c - DE)) * HWP;
  __shared__ float ssum[256], smax[256];
  float s = 0.f, m = -1e30f;
  for (int i = threadIdx.x; i < HWP; i += blockDim.x) {
    float v = base[i]; s += v; m = fmaxf(m, v);
  }
  ssum[threadIdx.x] = s; smax[threadIdx.x] = m; __syncthreads();
  for (int st = 128; st > 0; st >>= 1) {
    if ((int)threadIdx.x < st) {
      ssum[threadIdx.x] += ssum[threadIdx.x + st];
      smax[threadIdx.x] = fmaxf(smax[threadIdx.x], smax[threadIdx.x + st]);
    }
    __syncthreads();
  }
  if (threadIdx.x == 0) { avg[row] = ssum[0] / (float)HWP; mx[row] = smax[0]; }
}

// ca = sigmoid(fc2(relu(fc1(avg))) + fc2(relu(fc1(max))))   (one block per n)
__global__ void se_gate(const float* __restrict__ avg, const float* __restrict__ mx,
                        const float* __restrict__ fc1, const float* __restrict__ fc2,
                        float* __restrict__ ca)
{
  const int CT = DE + DI, HID = (DE + DI) / 16; // 192, 12
  int n = blockIdx.x, c = threadIdx.x;
  __shared__ float hsum[HID];
  if (c < HID) {
    float sa = 0.f, sm = 0.f;
    for (int i = 0; i < CT; ++i) {
      float w = fc1[c * CT + i];
      sa += w * avg[n * CT + i];
      sm += w * mx[n * CT + i];
    }
    hsum[c] = fmaxf(sa, 0.f) + fmaxf(sm, 0.f);   // fc2 linear -> sum hiddens
  }
  __syncthreads();
  float s = 0.f;
  for (int t = 0; t < HID; ++t) s += fc2[c * HID + t] * hsum[t];
  ca[n * CT + c] = 1.f / (1.f + expf(-s));
}

// per-pixel mean/max over channels of ev*ca and img*ca -> sa_in (n,4,H,W)
__global__ void spatial_stats(const float* __restrict__ ev, const float* __restrict__ img,
                              const float* __restrict__ ca, float* __restrict__ sain)
{
  long idx = (long)blockIdx.x * blockDim.x + threadIdx.x;
  if (idx >= (long)BATCH * HWP) return;
  int n = (int)(idx / HWP), p = (int)(idx % HWP);
  float s = 0.f, m = -1e30f;
  for (int c = 0; c < DE; ++c) {
    float v = ev[((size_t)n * DE + c) * HWP + p] * ca[n * (DE + DI) + c];
    s += v; m = fmaxf(m, v);
  }
  sain[((size_t)n * 4 + 0) * HWP + p] = s / (float)DE;
  sain[((size_t)n * 4 + 1) * HWP + p] = m;
  s = 0.f; m = -1e30f;
  for (int c = 0; c < DI; ++c) {
    float v = img[((size_t)n * DI + c) * HWP + p] * ca[n * (DE + DI) + DE + c];
    s += v; m = fmaxf(m, v);
  }
  sain[((size_t)n * 4 + 2) * HWP + p] = s / (float)DI;
  sain[((size_t)n * 4 + 3) * HWP + p] = m;
}

// 7x7 conv (4->2, pad 3, no bias) + sigmoid
__global__ void sconv7(const float* __restrict__ sain, const float* __restrict__ w,
                       float* __restrict__ sa)
{
  long idx = (long)blockIdx.x * blockDim.x + threadIdx.x;
  if (idx >= (long)BATCH * HWP) return;
  int n = (int)(idx / HWP), p = (int)(idx % HWP);
  int y = p / WWD, x = p % WWD;
  for (int oc = 0; oc < 2; ++oc) {
    float acc = 0.f;
    for (int ic = 0; ic < 4; ++ic) {
      const float* in = sain + ((size_t)n * 4 + ic) * HWP;
      for (int ky = 0; ky < 7; ++ky) {
        int yy = y + ky - 3; if (yy < 0 || yy >= HH) continue;
        for (int kx = 0; kx < 7; ++kx) {
          int xx = x + kx - 3; if (xx < 0 || xx >= WWD) continue;
          acc += in[yy * WWD + xx] * w[((oc * 4 + ic) * 7 + ky) * 7 + kx];
        }
      }
    }
    sa[((size_t)n * 2 + oc) * HWP + p] = 1.f / (1.f + expf(-acc));
  }
}

// ev_rec = ev*ca*sa0*gamma_ev + ev ; img_rec = img*ca*sa1*gamma_img + img
__global__ void recalib(const float* __restrict__ ev, const float* __restrict__ img,
                        const float* __restrict__ ca, const float* __restrict__ sa,
                        const float* __restrict__ gev, const float* __restrict__ gim,
                        float* __restrict__ evout, float* __restrict__ imgout)
{
  long idx = (long)blockIdx.x * blockDim.x + threadIdx.x;
  if (idx >= (long)BATCH * (DE + DI) * HWP) return;
  int n = (int)(idx / ((long)(DE + DI) * HWP));
  long rem = idx % ((long)(DE + DI) * HWP);
  int c = (int)(rem / HWP), p = (int)(rem % HWP);
  if (c < DE) {
    float v = ev[((size_t)n * DE + c) * HWP + p];
    evout[((size_t)n * DE + c) * HWP + p] =
        v * ca[n * (DE + DI) + c] * sa[((size_t)n * 2 + 0) * HWP + p] * gev[c] + v;
  } else {
    int ci = c - DE;
    float v = img[((size_t)n * DI + ci) * HWP + p];
    imgout[((size_t)n * DI + ci) * HWP + p] =
        v * ca[n * (DE + DI) + c] * sa[((size_t)n * 2 + 1) * HWP + p] * gim[ci] + v;
  }
}

// channels_first LayerNorm over DI channels per pixel (biased variance)
__global__ void ln_cf(const float* __restrict__ x, const float* __restrict__ w,
                      const float* __restrict__ b, float* __restrict__ y)
{
  long idx = (long)blockIdx.x * blockDim.x + threadIdx.x;
  if (idx >= (long)BATCH * HWP) return;
  int n = (int)(idx / HWP), p = (int)(idx % HWP);
  const float* base = x + (size_t)n * DI * HWP + p;
  float s = 0.f, sq = 0.f;
  for (int c = 0; c < DI; ++c) { float v = base[(size_t)c * HWP]; s += v; sq += v * v; }
  float u = s / (float)DI;
  float var = sq / (float)DI - u * u;
  float inv = rsqrtf(var + 1e-6f);
  float* yb = y + (size_t)n * DI * HWP + p;
  for (int c = 0; c < DI; ++c)
    yb[(size_t)c * HWP] = w[c] * ((base[(size_t)c * HWP] - u) * inv) + b[c];
}

// softmax over the spatial axis, in place; one block per (n,c) row
__global__ void softmax_spatial(float* __restrict__ x)
{
  float* base = x + (size_t)blockIdx.x * HWP;
  __shared__ float red[256];
  float m = -1e30f;
  for (int i = threadIdx.x; i < HWP; i += blockDim.x) m = fmaxf(m, base[i]);
  red[threadIdx.x] = m; __syncthreads();
  for (int st = 128; st > 0; st >>= 1) {
    if ((int)threadIdx.x < st) red[threadIdx.x] = fmaxf(red[threadIdx.x], red[threadIdx.x + st]);
    __syncthreads();
  }
  m = red[0]; __syncthreads();
  float s = 0.f;
  for (int i = threadIdx.x; i < HWP; i += blockDim.x) s += expf(base[i] - m);
  red[threadIdx.x] = s; __syncthreads();
  for (int st = 128; st > 0; st >>= 1) {
    if ((int)threadIdx.x < st) red[threadIdx.x] += red[threadIdx.x + st];
    __syncthreads();
  }
  float inv = 1.f / red[0];
  for (int i = threadIdx.x; i < HWP; i += blockDim.x) base[i] = expf(base[i] - m) * inv;
}

// softmax over the 32 head-channels per (n,h,p), in place
__global__ void softmax_head(float* __restrict__ q)
{
  long idx = (long)blockIdx.x * blockDim.x + threadIdx.x;
  if (idx >= (long)BATCH * HEADS * HWP) return;
  int n = (int)(idx / ((long)HEADS * HWP));
  long rem = idx % ((long)HEADS * HWP);
  int h = (int)(rem / HWP), p = (int)(rem % HWP);
  float* base = q + ((size_t)n * DI + h * DK) * HWP + p;
  float m = -1e30f;
  for (int k = 0; k < DK; ++k) m = fmaxf(m, base[(size_t)k * HWP]);
  float s = 0.f;
  for (int k = 0; k < DK; ++k) s += expf(base[(size_t)k * HWP] - m);
  float inv = 1.f / s;
  for (int k = 0; k < DK; ++k) base[(size_t)k * HWP] = expf(base[(size_t)k * HWP] - m) * inv;
}

// out = ev_p * softmax2(g)[0] + img_rec * softmax2(g)[1]
__global__ void gate_combine(const float* __restrict__ graw, const float* __restrict__ evp,
                             const float* __restrict__ imgrec, float* __restrict__ out)
{
  long idx = (long)blockIdx.x * blockDim.x + threadIdx.x;
  if (idx >= (long)BATCH * DI * HWP) return;
  int n = (int)(idx / ((long)DI * HWP));
  int p = (int)(idx % HWP);
  float g0 = graw[((size_t)n * 2 + 0) * HWP + p];
  float g1 = graw[((size_t)n * 2 + 1) * HWP + p];
  float mm = fmaxf(g0, g1);
  float e0 = expf(g0 - mm), e1 = expf(g1 - mm);
  float inv = 1.f / (e0 + e1);
  out[idx] = evp[idx] * (e0 * inv) + imgrec[idx] * (e1 * inv);
}

// depthwise 3x3 (pad 1) + bias + ReLU
__global__ void dwconv3(const float* __restrict__ x, const float* __restrict__ w,
                        const float* __restrict__ b, float* __restrict__ y)
{
  const int C = DI / 4;
  long idx = (long)blockIdx.x * blockDim.x + threadIdx.x;
  if (idx >= (long)BATCH * C * HWP) return;
  int n = (int)(idx / ((long)C * HWP));
  long rem = idx % ((long)C * HWP);
  int c = (int)(rem / HWP), p = (int)(rem % HWP);
  int y0 = p / WWD, x0 = p % WWD;
  const float* in = x + ((size_t)n * C + c) * HWP;
  float acc = 0.f;
  for (int ky = 0; ky < 3; ++ky) {
    int yy = y0 + ky - 1; if (yy < 0 || yy >= HH) continue;
    for (int kx = 0; kx < 3; ++kx) {
      int xx = x0 + kx - 1; if (xx < 0 || xx >= WWD) continue;
      acc += in[yy * WWD + xx] * w[(c * 3 + ky) * 3 + kx];
    }
  }
  acc += b[c];
  y[idx] = fmaxf(acc, 0.f);
}

// ---------------------------------------------------------------------------
#define EWGRID(total) dim3((unsigned)(((total) + 255) / 256)), dim3(256)
#define CONVGRID dim3((unsigned)(((long)BATCH * NPT + 3) / 4)), dim3(128)

extern "C" void kernel_launch(void* const* d_in, const int* in_sizes, int n_in,
                              void* d_out, int out_size, void* d_ws, size_t ws_size,
                              hipStream_t stream) {
  (void)in_sizes; (void)n_in; (void)out_size; (void)ws_size;
  const float* ev      = (const float*)d_in[0];
  const float* img     = (const float*)d_in[1];
  const float* fc1_w   = (const float*)d_in[2];
  const float* fc2_w   = (const float*)d_in[3];
  const float* sconv_w = (const float*)d_in[4];
  const float* gamma_ev = (const float*)d_in[5];
  const float* gamma_img = (const float*)d_in[6];
  const float* proj_w  = (const float*)d_in[7];
  const float* nev_w   = (const float*)d_in[8];
  const float* nev_b   = (const float*)d_in[9];
  const float* nimg_w  = (const float*)d_in[10];
  const float* nimg_b  = (const float*)d_in[11];
  const float* i2e_kw  = (const float*)d_in[12];
  const float* i2e_kb  = (const float*)d_in[13];
  const float* i2e_qw  = (const float*)d_in[14];
  const float* i2e_qb  = (const float*)d_in[15];
  const float* i2e_vw  = (const float*)d_in[16];
  const float* i2e_vb  = (const float*)d_in[17];
  const float* i2e_rw  = (const float*)d_in[18];
  const float* i2e_rb  = (const float*)d_in[19];
  const float* e2i_kw  = (const float*)d_in[20];
  const float* e2i_kb  = (const float*)d_in[21];
  const float* e2i_qw  = (const float*)d_in[22];
  const float* e2i_qb  = (const float*)d_in[23];
  const float* e2i_vw  = (const float*)d_in[24];
  const float* e2i_vb  = (const float*)d_in[25];
  const float* e2i_rw  = (const float*)d_in[26];
  const float* e2i_rb  = (const float*)d_in[27];
  const float* gate1_w = (const float*)d_in[28];
  const float* gate2_w = (const float*)d_in[29];
  const float* nffn_w  = (const float*)d_in[30];
  const float* nffn_b  = (const float*)d_in[31];
  const float* ffn1_w  = (const float*)d_in[32];
  const float* ffn1_b  = (const float*)d_in[33];
  const float* dw_w    = (const float*)d_in[34];
  const float* dw_b    = (const float*)d_in[35];
  const float* ffn2_w  = (const float*)d_in[36];
  const float* ffn2_b  = (const float*)d_in[37];

  float* out    = (float*)d_out;                         // (B,DI,H,W) final
  float* evout  = out + (size_t)BATCH * DI * HWP;        // (B,DE,H,W) = ev_rec
  float* imgout = evout + (size_t)BATCH * DE * HWP;      // (B,DI,H,W) = img_rec

  float* ws = (float*)d_ws;
  const size_t SZ = (size_t)BATCH * DI * HWP;            // 26.2M floats
  float* bufA = ws;            // ev_p
  float* bufB = ws + SZ;       // ev_f / evf2 / LN(out)
  float* bufC = ws + 2 * SZ;   // img_f / out
  float* bufD = ws + 3 * SZ;   // K / att / V2 / imgf2
  float* bufE = ws + 4 * SZ;   // Q / Q2
  float* bufF = ws + 5 * SZ;   // V / K2 / att2
  float* smallp = ws + 6 * SZ;
  float* part  = smallp;                                  // NCHUNK*CTXSZ
  float* ctxT  = part + (size_t)NCHUNK * CTXSZ;           // CTXSZ
  float* avg   = ctxT + CTXSZ;
  float* mxs   = avg + BATCH * (DE + DI);
  float* ca    = mxs + BATCH * (DE + DI);
  float* sain  = ca + BATCH * (DE + DI);                  // (B,4,H,W)
  float* sa    = sain + (size_t)BATCH * 4 * HWP;          // (B,2,H,W)
  float* gateh = sa + (size_t)BATCH * 2 * HWP;            // (B,16,H,W)
  float* graw  = gateh + (size_t)BATCH * 16 * HWP;        // (B,2,H,W)
  float* z1    = graw + (size_t)BATCH * 2 * HWP;          // (B,32,H,W)
  float* z2    = z1 + (size_t)BATCH * 32 * HWP;           // (B,32,H,W)

  // ---- channel SE gate ----------------------------------------------------
  channel_stats<<<dim3(BATCH * (DE + DI)), dim3(256), 0, stream>>>(ev, img, avg, mxs);
  se_gate<<<dim3(BATCH), dim3(DE + DI), 0, stream>>>(avg, mxs, fc1_w, fc2_w, ca);
  // ---- spatial gate + recalibration (ev_rec/img_rec -> d_out slices) ------
  spatial_stats<<<EWGRID((long)BATCH * HWP), 0, stream>>>(ev, img, ca, sain);
  sconv7<<<EWGRID((long)BATCH * HWP), 0, stream>>>(sain, sconv_w, sa);
  recalib<<<EWGRID((long)BATCH * (DE + DI) * HWP), 0, stream>>>(
      ev, img, ca, sa, gamma_ev, gamma_img, evout, imgout);
  // ---- projection + LayerNorms -------------------------------------------
  conv1x1_wmma<DE, DI, DE><<<CONVGRID, 0, stream>>>(
      evout, nullptr, proj_w, nullptr, nullptr, 0, bufA);
  ln_cf<<<EWGRID((long)BATCH * HWP), 0, stream>>>(bufA, nev_w, nev_b, bufB);
  ln_cf<<<EWGRID((long)BATCH * HWP), 0, stream>>>(imgout, nimg_w, nimg_b, bufC);
  // ---- i2e cross attention: evf2 = eca(ev_f, img_f) + ev_p ----------------
  conv1x1_wmma<DI, DI, DI><<<CONVGRID, 0, stream>>>(
      bufC, nullptr, i2e_kw, i2e_kb, nullptr, 0, bufD);
  softmax_spatial<<<dim3(BATCH * DI), dim3(256), 0, stream>>>(bufD);
  conv1x1_wmma<DI, DI, DI><<<CONVGRID, 0, stream>>>(
      bufB, nullptr, i2e_qw, i2e_qb, nullptr, 0, bufE);
  softmax_head<<<EWGRID((long)BATCH * HEADS * HWP), 0, stream>>>(bufE);
  conv1x1_wmma<DI, DI, DI><<<CONVGRID, 0, stream>>>(
      bufC, nullptr, i2e_vw, i2e_vb, nullptr, 0, bufF);
  ctx_wmma<<<dim3((BATCH * HEADS * NCHUNK + 3) / 4), dim3(128), 0, stream>>>(
      bufD, bufF, part);
  ctx_reduce<<<dim3((CTXSZ + 255) / 256), dim3(256), 0, stream>>>(part, ctxT);
  att_wmma<<<dim3((BATCH * HEADS * NPT + 3) / 4), dim3(128), 0, stream>>>(
      ctxT, bufE, bufD);
  conv1x1_wmma<DI, DI, DI><<<CONVGRID, 0, stream>>>(
      bufD, nullptr, i2e_rw, i2e_rb, bufA, 0, bufB);                  // evf2
  // ---- e2i cross attention: imgf2 = eca(img_f, evf2) + img_rec ------------
  conv1x1_wmma<DI, DI, DI><<<CONVGRID, 0, stream>>>(
      bufB, nullptr, e2i_kw, e2i_kb, nullptr, 0, bufF);
  softmax_spatial<<<dim3(BATCH * DI), dim3(256), 0, stream>>>(bufF);
  conv1x1_wmma<DI, DI, DI><<<CONVGRID, 0, stream>>>(
      bufC, nullptr, e2i_qw, e2i_qb, nullptr, 0, bufE);
  softmax_head<<<EWGRID((long)BATCH * HEADS * HWP), 0, stream>>>(bufE);
  conv1x1_wmma<DI, DI, DI><<<CONVGRID, 0, stream>>>(
      bufB, nullptr, e2i_vw, e2i_vb, nullptr, 0, bufD);
  ctx_wmma<<<dim3((BATCH * HEADS * NCHUNK + 3) / 4), dim3(128), 0, stream>>>(
      bufF, bufD, part);
  ctx_reduce<<<dim3((CTXSZ + 255) / 256), dim3(256), 0, stream>>>(part, ctxT);
  att_wmma<<<dim3((BATCH * HEADS * NPT + 3) / 4), dim3(128), 0, stream>>>(
      ctxT, bufE, bufF);
  conv1x1_wmma<DI, DI, DI><<<CONVGRID, 0, stream>>>(
      bufF, nullptr, e2i_rw, e2i_rb, imgout, 0, bufD);                // imgf2
  // ---- gate + combine -----------------------------------------------------
  conv1x1_wmma<2 * DI, 2 * DI / 16, DI><<<CONVGRID, 0, stream>>>(
      bufB, bufD, gate1_w, nullptr, nullptr, 1, gateh);
  conv1x1_wmma<16, 2, 16><<<CONVGRID, 0, stream>>>(
      gateh, nullptr, gate2_w, nullptr, nullptr, 0, graw);
  gate_combine<<<EWGRID((long)BATCH * DI * HWP), 0, stream>>>(graw, bufA, imgout, bufC);
  // ---- FFN ----------------------------------------------------------------
  ln_cf<<<EWGRID((long)BATCH * HWP), 0, stream>>>(bufC, nffn_w, nffn_b, bufB);
  conv1x1_wmma<DI, DI / 4, DI><<<CONVGRID, 0, stream>>>(
      bufB, nullptr, ffn1_w, ffn1_b, nullptr, 0, z1);
  dwconv3<<<EWGRID((long)BATCH * (DI / 4) * HWP), 0, stream>>>(z1, dw_w, dw_b, z2);
  conv1x1_wmma<DI / 4, DI, DI / 4><<<CONVGRID, 0, stream>>>(
      z2, nullptr, ffn2_w, ffn2_b, bufC, 0, out);
}
